// ModelTeacher_25142738551377
// MI455X (gfx1250) — matrix operand order repaired and checked by reading, not compiled, not run on hardware
//
#include <hip/hip_runtime.h>
#include <math.h>

#define N_RAYS   32768
#define N_PLANES 32
#define N_TILES  ((N_PLANES * N_RAYS) / 16)
#define WAVES    4
#define EPSV     1e-8f

typedef _Float16 v8h  __attribute__((ext_vector_type(8)));
typedef _Float16 v16h __attribute__((ext_vector_type(16)));
typedef float    v8f  __attribute__((ext_vector_type(8)));

// A fragment (16-bit A 16x32 layout): lane holds row m = lane&15,
// elements 0-15 = K[kk*32 + (lane>>4)*16 .. +15] (contiguous, matches B),
// where kbase = kk*32 + (lane>>4)*8.  Two contiguous 16B LDS loads.
__device__ __forceinline__ v16h load_a_frag(const _Float16* row, int kbase) {
    v8h lo = *(const v8h*)(row + kbase + (kbase & 8));
    v8h hi = *(const v8h*)(row + kbase + (kbase & 8) + 8);
    return __builtin_shufflevector(lo, hi, 0,1,2,3,4,5,6,7,8,9,10,11,12,13,14,15);
}

// fast sigmoid: single v_exp + single v_rcp (no IEEE div sequence)
__device__ __forceinline__ float sigmoidf_(float x) {
    return __builtin_amdgcn_rcpf(1.0f + __expf(-x));
}

__global__ __launch_bounds__(WAVES * 32, 1)
void mlp_kernel(const float* __restrict__ ndc,  const float* __restrict__ cam_pos,
                const float* __restrict__ cam_R, const float* __restrict__ basis,
                const float* __restrict__ center, const float* __restrict__ wh,
                const float* __restrict__ W0, const float* __restrict__ b0,
                const float* __restrict__ W1, const float* __restrict__ b1,
                const float* __restrict__ Wa, const float* __restrict__ ba,
                const float* __restrict__ Wc1, const float* __restrict__ bc1,
                const float* __restrict__ Wc2, const float* __restrict__ bc2,
                float* __restrict__ t_ws, float* __restrict__ rgba_ws)
{
    // Weights transposed in LDS as f16: sW*T[n][k].  Bias folded into pad K rows:
    //  - sW0T row 60 = b0 (A col 60 = 1.0)
    //  - sWc1T row 152 = bc1, sWaT row 152 col0 = ba (A col 216 = 1.0)
    __shared__ __attribute__((aligned(32))) _Float16 sW0T [128][64];   // 16 KB
    __shared__ __attribute__((aligned(32))) _Float16 sW1T [128][128];  // 32 KB
    __shared__ __attribute__((aligned(32))) _Float16 sWc1T[64][160];   // 20 KB
    __shared__ __attribute__((aligned(32))) _Float16 sWaT [16][160];   //  5 KB
    __shared__ __attribute__((aligned(32))) _Float16 sWc2T[16][64];    //  2 KB
    // Per-wave activation staging: cols 0..63 emb_pos (later hc), 64..191 h, 192..223 emb_dir(+pad)
    __shared__ __attribute__((aligned(32))) _Float16 stage[WAVES][16][224]; // 28 KB

    const int tid = threadIdx.x;

    for (int i = tid; i < 128 * 64; i += blockDim.x) {
        int nr = i >> 6, k = i & 63;
        _Float16 val = (_Float16)0.f;
        if (k < 60)       val = (_Float16)W0[k * 128 + nr];
        else if (k == 60) val = (_Float16)b0[nr];
        sW0T[nr][k] = val;
    }
    for (int i = tid; i < 128 * 128; i += blockDim.x) {
        int nr = i >> 7, k = i & 127;
        sW1T[nr][k] = (_Float16)W1[k * 128 + nr];
    }
    for (int i = tid; i < 64 * 160; i += blockDim.x) {
        int nr = i / 160, k = i % 160;
        _Float16 val = (_Float16)0.f;
        if (k < 152)       val = (_Float16)Wc1[k * 64 + nr];
        else if (k == 152) val = (_Float16)bc1[nr];
        sWc1T[nr][k] = val;
    }
    for (int i = tid; i < 16 * 160; i += blockDim.x) {
        int nr = i / 160, k = i % 160;
        _Float16 val = (_Float16)0.f;
        if (nr == 0) {
            if (k < 128)       val = (_Float16)Wa[k];
            else if (k == 152) val = (_Float16)ba[0];
        }
        sWaT[nr][k] = val;
    }
    for (int i = tid; i < 16 * 64; i += blockDim.x) {
        int nr = i >> 6, k = i & 63;
        sWc2T[nr][k] = (nr < 3) ? (_Float16)Wc2[k * 3 + nr] : (_Float16)0.f;
    }
    __syncthreads();

    const int lane = tid & 31;
    const int wv   = tid >> 5;
    const int nn   = lane & 15;   // column within n-tile; also sample id for geometry
    const int half = lane >> 4;

    float R[9];
    #pragma unroll
    for (int i = 0; i < 9; i++) R[i] = cam_R[i];
    const float o0 = cam_pos[0], o1 = cam_pos[1], o2 = cam_pos[2];

    // Per-lane bias slice for layer 1 (no pad room in K=128)
    float b1v[8];
    #pragma unroll
    for (int j = 0; j < 8; j++) b1v[j] = b1[j * 16 + nn];
    const float bc2l = (nn < 3) ? bc2[nn] : 0.f;

    _Float16* const srow = &stage[wv][nn][0];   // this lane's A row (m = nn)

    // Constant pad region (cols 216..223) never overwritten inside the loop:
    // col 216 = 1.0 feeds the bias rows of Wc1/Wa.  (duplicate same-value writes are fine)
    srow[216] = (_Float16)1.f;
    #pragma unroll
    for (int z = 217; z < 224; z++) srow[z] = (_Float16)0.f;

    const int wave_global = blockIdx.x * WAVES + wv;
    const int wave_stride = gridDim.x * WAVES;
    const v8f vzero = {0.f, 0.f, 0.f, 0.f, 0.f, 0.f, 0.f, 0.f};

    for (int tileI = wave_global; tileI < N_TILES; tileI += wave_stride) {
        const int s0 = tileI * 16;
        const int p  = s0 >> 15;          // s0 / N_RAYS
        const int n0 = s0 & (N_RAYS - 1);
        const int nray = n0 + nn;

        // ---- geometry for sample m = nn (all lanes compute; halves duplicate) ----
        float x0 = ndc[nray * 3 + 0], x1 = ndc[nray * 3 + 1], x2 = ndc[nray * 3 + 2];
        float d0 = R[0] * x0 + R[1] * x1 + R[2] * x2;
        float d1 = R[3] * x0 + R[4] * x1 + R[5] * x2;
        float d2 = R[6] * x0 + R[7] * x1 + R[8] * x2;
        float np0 = basis[p * 9 + 2], np1 = basis[p * 9 + 5], np2 = basis[p * 9 + 8];
        float c0 = center[p * 3 + 0], c1 = center[p * 3 + 1], c2 = center[p * 3 + 2];
        float hw0 = wh[p * 2 + 0] * 0.5f, hw1 = wh[p * 2 + 1] * 0.5f;
        float denom = np0 * d0 + np1 * d1 + np2 * d2;
        denom = (fabsf(denom) < EPSV) ? EPSV : denom;
        float num = (c0 - o0) * np0 + (c1 - o1) * np1 + (c2 - o2) * np2;
        float t = num * __builtin_amdgcn_rcpf(denom);          // single v_rcp_f32
        float wx = o0 + t * d0, wy = o1 + t * d1, wz = o2 + t * d2;
        float rx = wx - c0, ry = wy - c1, rz = wz - c2;
        float u = rx * basis[p * 9 + 0] + ry * basis[p * 9 + 3] + rz * basis[p * 9 + 6];
        float v = rx * basis[p * 9 + 1] + ry * basis[p * 9 + 4] + rz * basis[p * 9 + 7];
        bool hit = (fabsf(u) <= hw0) && (fabsf(v) <= hw1) && (t > 0.f);
        float hitf = hit ? 1.f : 0.f;
        wx *= hitf; wy *= hitf; wz *= hitf;
        float vdx = wx - o0, vdy = wy - o1, vdz = wz - o2;
        float inv = __builtin_amdgcn_rcpf(
            sqrtf(vdx * vdx + vdy * vdy + vdz * vdz) + EPSV);  // one rcp for 3 components
        vdx *= inv; vdy *= inv; vdz *= inv;

        // ---- harmonic encodings via double-angle recursion, split across wave halves ----
        // half 0 covers freqs 2^0..2^4 (pos) / 2^0..2^1 (dir); half 1 covers 2^5..2^9 / 2^2..2^3.
        {
            const float wc3[3] = {wx, wy, wz};
            const float f0p = (half == 0) ? 1.f : 32.f;
            #pragma unroll
            for (int c = 0; c < 3; c++) {
                float a = wc3[c] * f0p;
                float s = __sinf(a), co = __cosf(a);
                const int base = c * 20 + half * 5;
                #pragma unroll
                for (int i = 0; i < 5; i++) {
                    srow[base + i]      = (_Float16)s;
                    srow[base + 10 + i] = (_Float16)co;
                    float s2 = 2.f * s * co;
                    float c2 = co * co - s * s;
                    s = s2; co = c2;
                }
            }
            const float vc3[3] = {vdx, vdy, vdz};
            const float f0d = (half == 0) ? 1.f : 4.f;
            #pragma unroll
            for (int c = 0; c < 3; c++) {
                float a = vc3[c] * f0d;
                float s = __sinf(a), co = __cosf(a);
                const int base = 192 + c * 8 + half * 2;
                #pragma unroll
                for (int i = 0; i < 2; i++) {
                    srow[base + i]     = (_Float16)s;
                    srow[base + 4 + i] = (_Float16)co;
                    float s2 = 2.f * s * co;
                    float c2 = co * co - s * s;
                    s = s2; co = c2;
                }
            }
            // pad cols 60..63 are clobbered by hc staging each iter -> rewrite
            // (col 60 = 1.0 feeds the b0 bias row of sW0T); all lanes, no divergence
            srow[60] = (_Float16)1.f;
            srow[61] = (_Float16)0.f;
            srow[62] = (_Float16)0.f;
            srow[63] = (_Float16)0.f;
        }

        // ---- Layer 0: emb_pos(64) @ [W0;b0] -> h (128)  [16 WMMAs, bias folded] ----
        v8f acc[8];
        #pragma unroll
        for (int j = 0; j < 8; j++) acc[j] = vzero;
        #pragma unroll
        for (int kk = 0; kk < 2; kk++) {
            v16h afrag = load_a_frag(srow, kk * 32 + half * 8);
            #pragma unroll
            for (int j = 0; j < 8; j++) {
                v16h bfrag = *(const v16h*)(&sW0T[j * 16 + nn][kk * 32 + half * 16]);
                acc[j] = __builtin_amdgcn_wmma_f32_16x16x32_f16(
                    false, afrag, false, bfrag, (short)0, acc[j], false, false);
            }
        }
        #pragma unroll
        for (int j = 0; j < 8; j++) {
            #pragma unroll
            for (int r = 0; r < 8; r++) {
                float hv = acc[j][r];
                hv = hv > 0.f ? hv : 0.f;
                stage[wv][half * 8 + r][64 + j * 16 + nn] = (_Float16)hv;
            }
        }

        // ---- Layer 1: h(128) @ W1 + b1 -> h2 (128)  [32 WMMAs] ----
        v8f acc2[8];
        #pragma unroll
        for (int j = 0; j < 8; j++) acc2[j] = vzero;
        #pragma unroll
        for (int kk = 0; kk < 4; kk++) {
            v16h afrag = load_a_frag(srow + 64, kk * 32 + half * 8);
            #pragma unroll
            for (int j = 0; j < 8; j++) {
                v16h bfrag = *(const v16h*)(&sW1T[j * 16 + nn][kk * 32 + half * 16]);
                acc2[j] = __builtin_amdgcn_wmma_f32_16x16x32_f16(
                    false, afrag, false, bfrag, (short)0, acc2[j], false, false);
            }
        }
        #pragma unroll
        for (int j = 0; j < 8; j++) {
            #pragma unroll
            for (int r = 0; r < 8; r++) {
                float hv = acc2[j][r] + b1v[j];
                hv = hv > 0.f ? hv : 0.f;
                stage[wv][half * 8 + r][64 + j * 16 + nn] = (_Float16)hv;
            }
        }

        // ---- Layer c1 + alpha head: concat(h2, emb_dir, 1)(160) @ [Wc1;bc1 | Wa;ba]  [25 WMMAs] ----
        v8f acc3[4];
        #pragma unroll
        for (int j = 0; j < 4; j++) acc3[j] = vzero;
        v8f accA = vzero;   // alpha head: n=0 column = h2 . Wa + ba
        #pragma unroll
        for (int kk = 0; kk < 5; kk++) {
            v16h afrag = load_a_frag(srow + 64, kk * 32 + half * 8);
            #pragma unroll
            for (int j = 0; j < 4; j++) {
                v16h bfrag = *(const v16h*)(&sWc1T[j * 16 + nn][kk * 32 + half * 16]);
                acc3[j] = __builtin_amdgcn_wmma_f32_16x16x32_f16(
                    false, afrag, false, bfrag, (short)0, acc3[j], false, false);
            }
            v16h bfragA = *(const v16h*)(&sWaT[nn][kk * 32 + half * 16]);
            accA = __builtin_amdgcn_wmma_f32_16x16x32_f16(
                false, afrag, false, bfragA, (short)0, accA, false, false);
        }
        // stage hc = relu(acc3) into cols 0..63 (emb_pos region, no longer needed)
        #pragma unroll
        for (int j = 0; j < 4; j++) {
            #pragma unroll
            for (int r = 0; r < 8; r++) {
                float hv = acc3[j][r];
                hv = hv > 0.f ? hv : 0.f;
                stage[wv][half * 8 + r][j * 16 + nn] = (_Float16)hv;
            }
        }

        // ---- rgb head: hc(64) @ Wc2 -> 3 channels (cols 0..2)  [2 WMMAs] ----
        v8f accC = vzero;
        #pragma unroll
        for (int kk = 0; kk < 2; kk++) {
            v16h afrag = load_a_frag(srow, kk * 32 + half * 8);
            v16h bfrag = *(const v16h*)(&sWc2T[nn][kk * 32 + half * 16]);
            accC = __builtin_amdgcn_wmma_f32_16x16x32_f16(
                false, afrag, false, bfrag, (short)0, accC, false, false);
        }

        // ---- write t and rgba*hit ----
        float tm[8], hm[8], alphav[8], a3[8];
        #pragma unroll
        for (int r = 0; r < 8; r++) {
            int src = half * 8 + r;
            tm[r] = __shfl(t, src, 32);
            hm[r] = __shfl(hitf, src, 32);
        }
        #pragma unroll
        for (int r = 0; r < 8; r++) alphav[r] = sigmoidf_(accA[r]);  // ba folded in
        #pragma unroll
        for (int r = 0; r < 8; r++) a3[r] = __shfl(alphav[r], half * 16, 32); // from lane n=0
        if (nn == 0) {
            #pragma unroll
            for (int r = 0; r < 8; r++)
                t_ws[p * N_RAYS + n0 + half * 8 + r] = tm[r];
        }
        if (nn < 4) {
            #pragma unroll
            for (int r = 0; r < 8; r++) {
                int idx = p * N_RAYS + n0 + half * 8 + r;
                float valc = (nn < 3) ? sigmoidf_(accC[r] + bc2l) * hm[r]
                                      : a3[r] * hm[r];
                rgba_ws[idx * 4 + nn] = valc;
            }
        }
    }
}

__global__ __launch_bounds__(256)
void composite_kernel(const float* __restrict__ t_ws, const float* __restrict__ rgba_ws,
                      float* __restrict__ out)
{
    int n = blockIdx.x * blockDim.x + threadIdx.x;
    if (n >= N_RAYS) return;
    float tv[N_PLANES];
    int   pi[N_PLANES];
    for (int p = 0; p < N_PLANES; p++) { tv[p] = t_ws[p * N_RAYS + n]; pi[p] = p; }
    // stable insertion sort ascending in t (mirrors argsort)
    for (int i = 1; i < N_PLANES; i++) {
        float key = tv[i]; int kp = pi[i];
        int j = i - 1;
        while (j >= 0 && tv[j] > key) { tv[j + 1] = tv[j]; pi[j + 1] = pi[j]; j--; }
        tv[j + 1] = key; pi[j + 1] = kp;
    }
    float trans = 1.f, c0 = 0.f, c1 = 0.f, c2 = 0.f, dsum = 0.f, wsum = 0.f;
    for (int k = 0; k < N_PLANES; k++) {
        int p = pi[k];
        const float4 rg = *(const float4*)&rgba_ws[(p * N_RAYS + n) * 4];
        float a = rg.w;
        float w = a * trans;
        dsum += tv[k] * w;
        c0 += rg.x * w; c1 += rg.y * w; c2 += rg.z * w;
        wsum += w;
        trans *= (1.f - a);
    }
    float bg = 1.f - wsum;
    out[n * 4 + 0] = c0 + bg;
    out[n * 4 + 1] = c1 + bg;
    out[n * 4 + 2] = c2 + bg;
    out[n * 4 + 3] = dsum;
}

extern "C" void kernel_launch(void* const* d_in, const int* in_sizes, int n_in,
                              void* d_out, int out_size, void* d_ws, size_t ws_size,
                              hipStream_t stream) {
    const float* ndc  = (const float*)d_in[0];
    const float* cpos = (const float*)d_in[1];
    const float* cR   = (const float*)d_in[2];
    const float* bas  = (const float*)d_in[3];
    const float* cen  = (const float*)d_in[4];
    const float* whp  = (const float*)d_in[5];
    const float* W0   = (const float*)d_in[6];
    const float* b0   = (const float*)d_in[7];
    const float* W1   = (const float*)d_in[8];
    const float* b1   = (const float*)d_in[9];
    const float* Wa   = (const float*)d_in[10];
    const float* ba   = (const float*)d_in[11];
    const float* Wc1  = (const float*)d_in[12];
    const float* bc1  = (const float*)d_in[13];
    const float* Wc2  = (const float*)d_in[14];
    const float* bc2  = (const float*)d_in[15];

    float* t_ws    = (float*)d_ws;                     //  4 MB
    float* rgba_ws = t_ws + (size_t)N_PLANES * N_RAYS; // 16 MB

    mlp_kernel<<<2048, WAVES * 32, 0, stream>>>(
        ndc, cpos, cR, bas, cen, whp, W0, b0, W1, b1, Wa, ba, Wc1, bc1, Wc2, bc2,
        t_ws, rgba_ws);
    composite_kernel<<<N_RAYS / 256, 256, 0, stream>>>(t_ws, rgba_ws, (float*)d_out);
}
